// Graph_Discriminator_42494406427086
// MI455X (gfx1250) — matrix operand order repaired and checked by reading, not compiled, not run on hardware
//
#include <hip/hip_runtime.h>
#include <math.h>

// ---------------------------------------------------------------------------
// Graph discriminator (3 interaction-net layers) for MI455X / gfx1250.
// Compute-bound (~160 GFLOP vs ~20MB): edge MLP fe2 (128x128 over B*N*N pairs)
// runs on v_wmma_f32_16x16x32_bf16 with f32 accumulation.
// Round 2: spill-free edge kernel (A-tiles cached in VGPRs, B streamed from
// LDS inside a non-unrolled n-tile loop), branch-free lrelu = max(z, 0.2z),
// partial-tile masking only on the single partial j-tile.
// ---------------------------------------------------------------------------

typedef __bf16 bf16_t;
typedef __attribute__((ext_vector_type(16))) __bf16 v16bf;
typedef __attribute__((ext_vector_type(8)))  float  v8f;

#define NB    128     // batch
#define NH    100     // hits per graph
#define FE    128     // edge-net hidden width
#define ALPHA 0.2f    // leaky-relu slope
#define EWAVES 4      // waves (= nodes i) per edge-net block

// ---------------------------------------------------------------------------
// Pack fe2w [128(out) x 128(in)] into the WMMA B-operand (K=32 chunk) layout:
//   B = fe2w^T (KxN). For chunk kc, tile nt: lane L holds N = L%16;
//   lanes 0-15: K = kc*32+0..15 (vgpr g -> K=2g,2g+1); lanes 16-31: K=kc*32+16..31.
//   BP[((kc*8+nt)*32+lane)*16 + e] : 16 contiguous bf16 (32B) per lane.
// ---------------------------------------------------------------------------
__global__ void pack_fe2w(const float* __restrict__ fe2w, bf16_t* __restrict__ bp) {
    int tid = blockIdx.x * blockDim.x + threadIdx.x;   // 8192 = kc(4)*nt(8)*lane(32)*g(8)
    if (tid >= 4 * 8 * 32 * 8) return;
    int g    = tid & 7;
    int lane = (tid >> 3) & 31;
    int nt   = (tid >> 8) & 7;
    int kc   = tid >> 11;
    int n  = nt * 16 + (lane & 15);
    int k0 = kc * 32 + ((lane < 16) ? 0 : 16) + 2 * g;
    bf16_t* dst = bp + (((kc * 8 + nt) * 32 + lane) * 16) + 2 * g;
    dst[0] = (bf16_t)fe2w[n * FE + k0];
    dst[1] = (bf16_t)fe2w[n * FE + k0 + 1];
}

// ---------------------------------------------------------------------------
// One pass over NJT j-tiles starting at JT0: build bf16 A tiles in registers,
// then for each n-tile stream B from LDS, 4*NJT chained WMMAs, bias + lrelu +
// row-sum, lane-half reduce, accumulate into sAcc.
// ---------------------------------------------------------------------------
template <int NJT, int JT0, bool ACCUM>
__device__ __forceinline__ void edge_pass(const float* __restrict__ sU_w,
                                          const float* __restrict__ sV,
                                          const float* __restrict__ sCo,
                                          const float* __restrict__ sWn,
                                          const float* __restrict__ sB2,
                                          const bf16_t* __restrict__ sB,
                                          float* __restrict__ sAcc_w,
                                          float xi0, float xi1, int lane)
{
    const int off = (lane < 16) ? 0 : 8;   // A-operand K split per lane half
    const int jj  = lane & 15;             // A row / B column within lane

    union AUn { v16bf v; bf16_t e[16]; };
    AUn A[NJT][4];

#pragma unroll
    for (int jtl = 0; jtl < NJT; ++jtl) {
        const int  jt       = JT0 + jtl;
        const bool fullTile = (jt * 16 + 15) < NH;     // compile-time
        int  j      = jt * 16 + jj;
        bool jvalid = fullTile || (j < NH);
        int  jc     = fullTile ? j : (jvalid ? j : 0); // clamp partial tile
        float dx = sCo[jc * 2] - xi0, dy = sCo[jc * 2 + 1] - xi1;
        float sq  = dx * dx + dy * dy;
        float nrm = (sq > 0.f) ? sqrtf(sq) : 0.f;
        const float* vrow = sV + jc * FE;
#pragma unroll
        for (int kc = 0; kc < 4; ++kc) {
#pragma unroll
            for (int e = 0; e < 16; ++e) {
                // A layout: lanes 0-15 -> K=kc*32+0..15, lanes 16-31 -> K=kc*32+16..31
                int k = kc * 32 + 2 * off + e;
                float z  = sU_w[k] + vrow[k] + sWn[k] * nrm;
                float hv = fmaxf(z, ALPHA * z);        // leaky relu, branch-free
                if (!fullTile) hv = jvalid ? hv : 0.f; // zero padded rows
                A[jtl][kc].e[e] = (bf16_t)hv;
            }
        }
    }

#pragma unroll 1   // keep as a real loop: B stays an LDS stream, no LICM blowup
    for (int nt = 0; nt < 8; ++nt) {
        v16bf Bm[4];
#pragma unroll
        for (int kc = 0; kc < 4; ++kc)
            Bm[kc] = *(const v16bf*)(sB + ((kc * 8 + nt) * 32 + lane) * 16);
        float fb2 = sB2[nt * 16 + jj];
        float acc = 0.f;
#pragma unroll
        for (int jtl = 0; jtl < NJT; ++jtl) {
            const int  jt       = JT0 + jtl;
            const bool fullTile = (jt * 16 + 15) < NH;
            v8f c = {0.f, 0.f, 0.f, 0.f, 0.f, 0.f, 0.f, 0.f};
#pragma unroll
            for (int kc = 0; kc < 4; ++kc)
                c = __builtin_amdgcn_wmma_f32_16x16x32_bf16(
                        false, A[jtl][kc].v, false, Bm[kc], (short)0, c, false, false);
#pragma unroll
            for (int r = 0; r < 8; ++r) {
                // D layout: vgpr r -> M=r (lanes 0-15) / M=8+r (lanes 16-31)
                float z = c[r] + fb2;
                float l = fmaxf(z, ALPHA * z);
                if (fullTile) {
                    acc += l;
                } else {
                    int jrow = jt * 16 + r + ((lane < 16) ? 0 : 8);
                    acc += (jrow < NH) ? l : 0.f;
                }
            }
        }
        // combine the two lane halves (same n, complementary row sets)
        float tot = acc + __shfl_xor(acc, 16, 32);
        if (lane < 16) {
            if (ACCUM) sAcc_w[nt * 16 + lane] += tot;
            else       sAcc_w[nt * 16 + lane]  = tot;
        }
    }
}

// ---------------------------------------------------------------------------
// Edge network. Grid: (25 i-tiles, NB batches), 128 threads = 4 waves.
// Wave w owns node i = itile*4 + w (exactly covers NH=100; no idle waves).
// ---------------------------------------------------------------------------
template <int D>
__global__ __launch_bounds__(128, 1)
void edge_net(const float* __restrict__ xg,    // [NB,NH,D]
              const float* __restrict__ fe1w,  // [128, 2D+1]
              const float* __restrict__ fe1b,  // [128]
              const float* __restrict__ fe2b,  // [128]
              const bf16_t* __restrict__ bp,   // packed fe2w
              float* __restrict__ av)          // [NB,NH,128]
{
    extern __shared__ char dsm[];
    float*  sV   = (float*)dsm;               // NH*FE : v_j (right half of fe1)
    float*  sCo  = sV + NH * FE;              // NH*2  : coords for norms
    float*  sWn  = sCo + NH * 2;              // FE    : fe1w[:,2D] (norm column)
    float*  sB2  = sWn + FE;                  // FE    : fe2b
    float*  sU   = sB2 + FE;                  // EWAVES*FE : per-wave u_i + fe1b
    float*  sAcc = sU + EWAVES * FE;          // EWAVES*FE : per-wave av accumulator
    bf16_t* sB   = (bf16_t*)(sAcc + EWAVES * FE); // 4*8*32*16 bf16 packed fe2w^T

    const int b     = blockIdx.y;
    const int itile = blockIdx.x;
    const int tid   = threadIdx.x;
    const int lane  = tid & 31;
    const int wave  = tid >> 5;
    const int FI    = 2 * D + 1;

    // ---- stage V[j,k] = fe1w[k, D:2D] . x[b,j,:]  (thread: fixed k, j walks)
    for (int idx = tid; idx < NH * FE; idx += 128) {
        int j = idx >> 7, k = idx & 127;
        const float* xr = xg + (b * NH + j) * D;
        const float* wr = fe1w + k * FI + D;
        float acc = 0.f;
#pragma unroll
        for (int c = 0; c < D; ++c) acc += wr[c] * xr[c];
        sV[idx] = acc;
    }
    for (int idx = tid; idx < NH * 2; idx += 128)
        sCo[idx] = xg[(b * NH + (idx >> 1)) * D + (idx & 1)];
    for (int idx = tid; idx < FE; idx += 128) {
        sWn[idx] = fe1w[idx * FI + 2 * D];
        sB2[idx] = fe2b[idx];
    }
    {   // packed fe2w: 32KB copied as dword4
        const uint4* src = (const uint4*)bp;
        uint4*       dst = (uint4*)sB;
        for (int idx = tid; idx < 4 * 8 * 32 * 8 / 4; idx += 128)
            dst[idx] = src[idx];
    }

    const int i = itile * EWAVES + wave;      // always < NH
    __syncthreads();

    // ---- per-wave u_i[k] + fe1b[k] ---------------------------------------
    {
        const float* xr = xg + (b * NH + i) * D;
#pragma unroll
        for (int t = 0; t < 4; ++t) {
            int k = lane * 4 + t;
            const float* wr = fe1w + k * FI;
            float acc = fe1b[k];
#pragma unroll
            for (int c = 0; c < D; ++c) acc += wr[c] * xr[c];
            sU[wave * FE + k] = acc;
        }
    }
    __syncthreads();

    const float xi0 = sCo[i * 2], xi1 = sCo[i * 2 + 1];
    const float* sU_w   = sU + wave * FE;
    float*       sAcc_w = sAcc + wave * FE;

    // 7 j-tiles in two passes: A cache = NJT*4*8 = 128 VGPRs max
    edge_pass<4, 0, false>(sU_w, sV, sCo, sWn, sB2, sB, sAcc_w, xi0, xi1, lane);
    edge_pass<3, 4, true >(sU_w, sV, sCo, sWn, sB2, sB, sAcc_w, xi0, xi1, lane);

    // ---- write av[b,i,:] --------------------------------------------------
    for (int idx = lane; idx < FE; idx += 32)
        av[(b * NH + i) * FE + idx] = sAcc_w[idx];
}

// ---------------------------------------------------------------------------
// Node network: y = tanh(fn2 . tanh(fn1 . [av, x] + b1) + b2). Tiny (0.5 GFLOP
// total) -> plain f32 VALU. 8 waves/block, one node per wave; 1600 blocks.
// ---------------------------------------------------------------------------
template <int D, int OUTC, bool LAST>
__global__ void node_net(const float* __restrict__ av,
                         const float* __restrict__ xcur,
                         const float* __restrict__ fn1w,  // [128, 128+D]
                         const float* __restrict__ fn1b,
                         const float* __restrict__ fn2w,  // [OUTC, 128]
                         const float* __restrict__ fn2b,
                         float* __restrict__ outbuf)      // LAST ? [P] : [P, 2+OUTC]
{
    __shared__ float st[8][FE + D];
    __shared__ float sh[8][FE];
    const int tid = threadIdx.x, lane = tid & 31, wave = tid >> 5;
    const int p = blockIdx.x * 8 + wave;     // always < 12800 (exact grid)
    const int KD = FE + D;

    for (int idx = lane; idx < FE; idx += 32) st[wave][idx] = av[p * FE + idx];
    for (int idx = lane; idx < D;  idx += 32) st[wave][FE + idx] = xcur[p * D + idx];
    __syncthreads();

    for (int m = lane; m < FE; m += 32) {
        float acc = fn1b[m];
        const float* wr = fn1w + m * KD;
        for (int k = 0; k < KD; ++k) acc += wr[k] * st[wave][k];
        sh[wave][m] = tanhf(acc);
    }
    __syncthreads();

    for (int o = lane; o < OUTC; o += 32) {
        float acc = fn2b[o];
        const float* wr = fn2w + o * FE;
        for (int k = 0; k < FE; ++k) acc += wr[k] * sh[wave][k];
        float y = tanhf(acc);
        if (LAST) outbuf[p] = y;
        else      outbuf[p * (2 + OUTC) + 2 + o] = y;
    }
    if (!LAST && lane < 2) outbuf[p * (2 + OUTC) + lane] = xcur[p * D + lane];
}

// ---------------------------------------------------------------------------
__global__ void reduce_sigmoid(const float* __restrict__ sc, float* __restrict__ out) {
    int b = blockIdx.x * blockDim.x + threadIdx.x;
    if (b < NB) {
        float s = 0.f;
        for (int i = 0; i < NH; ++i) s += sc[b * NH + i];
        s *= (1.0f / NH);
        out[b] = 1.f / (1.f + expf(-s));
    }
}

// ---------------------------------------------------------------------------
extern "C" void kernel_launch(void* const* d_in, const int* in_sizes, int n_in,
                              void* d_out, int out_size, void* d_ws, size_t ws_size,
                              hipStream_t stream) {
    (void)in_sizes; (void)n_in; (void)out_size; (void)ws_size;

    const float* x0 = (const float*)d_in[0];
    auto P = [&](int layer, int idx) { return (const float*)d_in[1 + layer * 8 + idx]; };

    // workspace: AV (6.55MB) | X1 (0.82MB) | X2 (1.64MB) | scores | packed fe2w
    float*  AV = (float*)d_ws;
    float*  X1 = AV + NB * NH * FE;
    float*  X2 = X1 + NB * NH * 16;
    float*  SC = X2 + NB * NH * 32;
    bf16_t* BP = (bf16_t*)(SC + NB * NH);

    dim3 eg(25, NB);   // 25 * 4 waves = exactly NH nodes
    const size_t shmem =
        (size_t)((NH * FE + NH * 2 + FE + FE + 2 * EWAVES * FE) * 4
                 + 4 * 8 * 32 * 16 * 2);   // ~90KB dynamic LDS

    // ---- layer 0 (D=2) ----
    pack_fe2w<<<32, 256, 0, stream>>>(P(0, 2), BP);
    edge_net<2><<<eg, 128, shmem, stream>>>(x0, P(0, 0), P(0, 1), P(0, 3), BP, AV);
    node_net<2, 14, false><<<1600, 256, 0, stream>>>(AV, x0, P(0, 4), P(0, 5), P(0, 6), P(0, 7), X1);
    // ---- layer 1 (D=16) ----
    pack_fe2w<<<32, 256, 0, stream>>>(P(1, 2), BP);
    edge_net<16><<<eg, 128, shmem, stream>>>(X1, P(1, 0), P(1, 1), P(1, 3), BP, AV);
    node_net<16, 30, false><<<1600, 256, 0, stream>>>(AV, X1, P(1, 4), P(1, 5), P(1, 6), P(1, 7), X2);
    // ---- layer 2 (D=32) ----
    pack_fe2w<<<32, 256, 0, stream>>>(P(2, 2), BP);
    edge_net<32><<<eg, 128, shmem, stream>>>(X2, P(2, 0), P(2, 1), P(2, 3), BP, AV);
    node_net<32, 1, true><<<1600, 256, 0, stream>>>(AV, X2, P(2, 4), P(2, 5), P(2, 6), P(2, 7), SC);

    reduce_sigmoid<<<1, 128, 0, stream>>>(SC, (float*)d_out);
}
